// Net_60078002536518
// MI455X (gfx1250) — matrix operand order repaired and checked by reading, not compiled, not run on hardware
//
#include <hip/hip_runtime.h>

// ---------------------------------------------------------------------------
// Types for CDNA5 WMMA (wave32)
// ---------------------------------------------------------------------------
typedef __bf16 bf16_t;
typedef __attribute__((ext_vector_type(16))) __bf16 v16bf;
typedef __attribute__((ext_vector_type(8)))  float  v8f;
typedef __attribute__((ext_vector_type(8)))  unsigned int v8u;

#define NNODES   65536      // B*N
#define NEDGES   524288
#define BGRAPH   128
#define NPG      512        // nodes per graph
#define DDIM     128
#define RATIO    128
#define HEADS    4
#define NND      ((size_t)NNODES * DDIM)   // 8388608
#define PND      ((size_t)BGRAPH * RATIO * DDIM) // 2097152

// fp32 -> bf16 round-to-nearest-even
static __device__ inline unsigned short f2bfu(float f) {
  unsigned u = __builtin_bit_cast(unsigned, f);
  u = u + 0x7FFFu + ((u >> 16) & 1u);
  return (unsigned short)(u >> 16);
}
static __device__ inline __bf16 f2bf(float f) {
  unsigned short s = f2bfu(f);
  return __builtin_bit_cast(__bf16, s);
}

// ---------------------------------------------------------------------------
// Elementwise helpers
// ---------------------------------------------------------------------------
__global__ void k_zero(float* p, int count) {
  int i = blockIdx.x * 256 + threadIdx.x;
  if (i < count) p[i] = 0.0f;
}

__global__ void k_f32_to_bf16(const float* in, unsigned short* out, int count) {
  int i = blockIdx.x * 256 + threadIdx.x;
  if (i < count) out[i] = f2bfu(in[i]);
}

// W [K rows, N cols] fp32 row-major  ->  WT bf16 [N][K]
__global__ void k_transpose_bf16(const float* W, unsigned short* WT, int K, int N) {
  int i = blockIdx.x * 256 + threadIdx.x;
  if (i >= K * N) return;
  int kk = i / N, nn = i % N;
  WT[nn * K + kk] = f2bfu(W[i]);
}

// ---------------------------------------------------------------------------
// WMMA GEMM: out[rows,128] = f(Ab[rows,128] @ W[128,128] + bias) (+ residual)
// Ab bf16 row-major; WT bf16 transposed [col][k].
// mode: 0 = none, 1 = relu, 2 = residual + relu(gemm+bias)
// grid = rows/128, block = 256 (8 waves, each wave -> 16 rows x 128 cols)
// ---------------------------------------------------------------------------
__global__ __launch_bounds__(256) void k_gemm128(
    const unsigned short* __restrict__ Ab, const unsigned short* __restrict__ WT,
    const float* __restrict__ bias, const float* __restrict__ residual,
    float* __restrict__ out, int mode)
{
  int wave = threadIdx.x >> 5;
  int lane = threadIdx.x & 31;
  int rlo  = lane & 15;
  int half = lane >> 4;
  int row0 = blockIdx.x * 128 + wave * 16;
  const unsigned short* abase = Ab + (size_t)(row0 + rlo) * 128;

  v16bf afrag[4];
#pragma unroll
  for (int kc = 0; kc < 4; kc++) {
    const uint4* p0 = (const uint4*)(abase + kc * 32 + half * 16);
    const uint4* p1 = (const uint4*)(abase + kc * 32 + half * 16 + 8);
    uint4 lo = *p0, hi = *p1;
    v8u bits = {lo.x, lo.y, lo.z, lo.w, hi.x, hi.y, hi.z, hi.w};
    afrag[kc] = __builtin_bit_cast(v16bf, bits);
  }

#pragma unroll
  for (int ct = 0; ct < 8; ct++) {
    int col = ct * 16 + rlo;
    v8f c = {0.f, 0.f, 0.f, 0.f, 0.f, 0.f, 0.f, 0.f};
#pragma unroll
    for (int kc = 0; kc < 4; kc++) {
      const uint4* q0 = (const uint4*)(WT + (size_t)col * 128 + kc * 32 + half * 16);
      uint4 lo = q0[0], hi = q0[1];
      v8u bits = {lo.x, lo.y, lo.z, lo.w, hi.x, hi.y, hi.z, hi.w};
      v16bf bfrag = __builtin_bit_cast(v16bf, bits);
      c = __builtin_amdgcn_wmma_f32_16x16x32_bf16(false, afrag[kc], false, bfrag,
                                                  (short)0, c, false, false);
    }
    float bv = bias ? bias[col] : 0.0f;
#pragma unroll
    for (int r = 0; r < 8; r++) {
      int orow = row0 + half * 8 + r;
      float v = c[r] + bv;
      if (mode >= 1) v = v > 0.f ? v : 0.f;
      if (mode == 2) v += residual[(size_t)orow * 128 + col];
      out[(size_t)orow * 128 + col] = v;
    }
  }
}

// ---------------------------------------------------------------------------
// Graph kernels
// ---------------------------------------------------------------------------
__global__ void k_deg(const int* __restrict__ dst, float* deg, int E) {
  int e = blockIdx.x * 256 + threadIdx.x;
  if (e < E) atomicAdd(&deg[dst[e]], 1.0f);
}

__global__ void k_dinv(const float* deg, float* dinv, int n) {
  int i = blockIdx.x * 256 + threadIdx.x;
  if (i < n) dinv[i] = rsqrtf(deg[i] + 1.0f);
}

// out[dst] += h[src]*norm   (grid = E blocks, block = 128)
__global__ void k_scatter128(const float* __restrict__ h, const int* __restrict__ src,
                             const int* __restrict__ dst, const float* __restrict__ dinv,
                             float* out)
{
  int e = blockIdx.x, d = threadIdx.x;
  int s = src[e], t = dst[e];
  float nm = dinv[s] * dinv[t];
  atomicAdd(&out[(size_t)t * 128 + d], h[(size_t)s * 128 + d] * nm);
}

// accrelu=1: out += relu(scat + h*selfnorm + b); accrelu=0: out = scat + h*selfnorm + b
__global__ void k_gcn_finalize(const float* __restrict__ scat, const float* __restrict__ h,
                               const float* __restrict__ dinv, const float* __restrict__ b,
                               float* out, int accrelu)
{
  int i = blockIdx.x * 256 + threadIdx.x;   // i < NNODES*128
  int node = i >> 7, col = i & 127;
  float sn = dinv[node] * dinv[node];
  float v = scat[i] + h[i] * sn + b[col];
  if (accrelu) out[i] += fmaxf(v, 0.0f);
  else         out[i] = v;
}

// per-node dual dot product (one wave per node)
__global__ void k_dot2(const float* __restrict__ x, const float* __restrict__ w1,
                       const float* __restrict__ w2, float* s1, float* hv, int n)
{
  int gw = (blockIdx.x * blockDim.x + threadIdx.x) >> 5;
  int lane = threadIdx.x & 31;
  if (gw >= n) return;
  float a = 0.f, b = 0.f;
  for (int j = lane; j < 128; j += 32) {
    float xv = x[(size_t)gw * 128 + j];
    a += xv * w1[j];
    b += xv * w2[j];
  }
#pragma unroll
  for (int o = 16; o; o >>= 1) { a += __shfl_down(a, o, 32); b += __shfl_down(b, o, 32); }
  if (lane == 0) { s1[gw] = a; hv[gw] = b; }
}

__global__ void k_scatter1(const float* __restrict__ hv, const int* __restrict__ src,
                           const int* __restrict__ dst, const float* __restrict__ dinv,
                           float* s2, int E)
{
  int e = blockIdx.x * 256 + threadIdx.x;
  if (e >= E) return;
  int s = src[e], t = dst[e];
  atomicAdd(&s2[t], hv[s] * dinv[s] * dinv[t]);
}

__global__ void k_gla_combine(const float* s1, const float* hv, const float* s2,
                              const float* dinv, const float* b1, const float* b2,
                              float* out, int n)
{
  int i = blockIdx.x * 256 + threadIdx.x;
  if (i >= n) return;
  float sn = dinv[i] * dinv[i];
  out[i] = 0.5f * (s1[i] + b1[0]) + 0.5f * (s2[i] + hv[i] * sn + b2[0]);
}

// xf = softmax(scA,scB,scC) . (x1,x2,x3)
__global__ void k_mix3(const float* sA, const float* sB, const float* sC,
                       const float* x1, const float* x2, const float* x3, float* xf)
{
  int i = blockIdx.x * 256 + threadIdx.x;   // < NNODES*128
  int node = i >> 7;
  float a = sA[node], b = sB[node], c = sC[node];
  float m = fmaxf(a, fmaxf(b, c));
  float ea = expf(a - m), eb = expf(b - m), ec = expf(c - m);
  float inv = 1.0f / (ea + eb + ec);
  xf[i] = (ea * x1[i] + eb * x2[i] + ec * x3[i]) * inv;
}

// top-128 of 512 per graph, descending, jax tie-break (lower index first)
__global__ __launch_bounds__(512) void k_topk(const float* __restrict__ score,
                                              int* idx, float* vals)
{
  __shared__ float sv[512];
  __shared__ int   si[512];
  int b = blockIdx.x, t = threadIdx.x;
  sv[t] = score[b * 512 + t];
  si[t] = t;
  __syncthreads();
  for (int k = 2; k <= 512; k <<= 1) {
    for (int j = k >> 1; j > 0; j >>= 1) {
      int ixj = t ^ j;
      if (ixj > t) {
        bool up = ((t & k) == 0);
        float v0 = sv[t], v1 = sv[ixj];
        int   i0 = si[t], i1 = si[ixj];
        bool pre = (v0 > v1) || (v0 == v1 && i0 < i1);
        if (up ? !pre : pre) { sv[t] = v1; si[t] = i1; sv[ixj] = v0; si[ixj] = i0; }
      }
      __syncthreads();
    }
  }
  if (t < 128) { idx[b * 128 + t] = si[t]; vals[b * 128 + t] = sv[t]; }
}

// xp (bf16) = xf[gathered] * tanh(vals)
__global__ void k_gather_pool(const float* __restrict__ xf, const int* __restrict__ idx,
                              const float* __restrict__ vals, unsigned short* xpb)
{
  int br = blockIdx.x, d = threadIdx.x;   // grid B*128, block 128
  int b = br >> 7;
  int node = b * 512 + idx[br];
  float v = xf[(size_t)node * 128 + d] * tanhf(vals[br]);
  xpb[(size_t)br * 128 + d] = f2bfu(v);
}

// ---------------------------------------------------------------------------
// Attention: one wave per (b, h, 16-row q-tile).  grid = B*H*8, block = 32.
// att tile (16 x 512) fp32 in LDS.  O = Q + softmax(QK^T/sqrt(D)) V
// ---------------------------------------------------------------------------
__global__ __launch_bounds__(32) void k_attention(const float* __restrict__ Q,
                                                  const unsigned short* __restrict__ Kb,
                                                  const unsigned short* __restrict__ Vb,
                                                  float* __restrict__ O)
{
  __shared__ float att[16 * 512];
  int bw   = blockIdx.x;
  int tile = bw & 7;
  int h    = (bw >> 3) & 3;
  int b    = bw >> 5;
  int lane = threadIdx.x;
  int rlo  = lane & 15;
  int half = lane >> 4;
  int q0   = tile * 16;
  const float scale = 0.0883883476483184f;   // 1/sqrt(128)

  // A-frag from Q rows (fp32 -> bf16)
  v16bf aq;
  {
    const float* qbase = Q + (size_t)(b * 128 + q0 + rlo) * 128 + h * 32;
#pragma unroll
    for (int i = 0; i < 16; i++) {
      int kk = (half << 4) + i;
      aq[i] = f2bf(qbase[kk]);
    }
  }

  // Phase 1: scores
  for (int nt = 0; nt < 32; nt++) {
    const uint4* p = (const uint4*)(Kb + (size_t)(b * 512 + nt * 16 + rlo) * 128 + h * 32 + half * 16);
    uint4 lo = p[0], hi = p[1];
    v8u bits = {lo.x, lo.y, lo.z, lo.w, hi.x, hi.y, hi.z, hi.w};
    v16bf bk = __builtin_bit_cast(v16bf, bits);
    v8f c = {0.f, 0.f, 0.f, 0.f, 0.f, 0.f, 0.f, 0.f};
    c = __builtin_amdgcn_wmma_f32_16x16x32_bf16(false, aq, false, bk, (short)0, c, false, false);
#pragma unroll
    for (int r = 0; r < 8; r++)
      att[(half * 8 + r) * 512 + nt * 16 + rlo] = c[r] * scale;
  }
  __syncthreads();

  // Phase 2: row softmax (wave-local; each lane owns cols j == lane mod 32)
  for (int r = 0; r < 16; r++) {
    float* rowp = att + r * 512;
    float m = -1e30f;
    for (int j = lane; j < 512; j += 32) m = fmaxf(m, rowp[j]);
#pragma unroll
    for (int o = 16; o; o >>= 1) m = fmaxf(m, __shfl_xor(m, o, 32));
    float s = 0.f;
    for (int j = lane; j < 512; j += 32) { float e = expf(rowp[j] - m); rowp[j] = e; s += e; }
#pragma unroll
    for (int o = 16; o; o >>= 1) s += __shfl_xor(s, o, 32);
    float inv = 1.0f / s;
    for (int j = lane; j < 512; j += 32) rowp[j] *= inv;
  }
  __syncthreads();

  // Phase 3: O = Q + A*V
#pragma unroll
  for (int nt = 0; nt < 2; nt++) {
    v8f c;
#pragma unroll
    for (int r = 0; r < 8; r++)
      c[r] = Q[(size_t)(b * 128 + q0 + half * 8 + r) * 128 + h * 32 + nt * 16 + rlo];
    for (int kc = 0; kc < 16; kc++) {
      v16bf aa;
      const float* arow = att + rlo * 512 + kc * 32;
#pragma unroll
      for (int i = 0; i < 16; i++) {
        int kk = (half << 4) + i;
        aa[i] = f2bf(arow[kk]);
      }
      v16bf bv;
#pragma unroll
      for (int i = 0; i < 16; i++) {
        unsigned short uv = Vb[(size_t)(b * 512 + kc * 32 + half * 16 + i) * 128 + h * 32 + nt * 16 + rlo];
        bv[i] = __builtin_bit_cast(__bf16, uv);
      }
      c = __builtin_amdgcn_wmma_f32_16x16x32_bf16(false, aa, false, bv, (short)0, c, false, false);
    }
#pragma unroll
    for (int r = 0; r < 8; r++)
      O[(size_t)(b * 128 + q0 + half * 8 + r) * 128 + h * 32 + nt * 16 + rlo] = c[r];
  }
}

// ---------------------------------------------------------------------------
// Readout / classifier tail
// ---------------------------------------------------------------------------
__global__ void k_readout(const float* __restrict__ O2, const float* __restrict__ ro_w,
                          const float* __restrict__ ro_b, float* out_xg, unsigned short* xgb)
{
  int i = blockIdx.x * 256 + threadIdx.x;   // B*128
  if (i >= 128 * 128) return;
  int b = i >> 7, d = i & 127;
  float s = ro_b[0];
  for (int r = 0; r < 128; r++) s += ro_w[r] * O2[(size_t)(b * 128 + r) * 128 + d];
  out_xg[i] = s;
  xgb[i] = f2bfu(s);
}

__global__ void k_logits(const float* __restrict__ hc, const float* __restrict__ c2w,
                         const float* __restrict__ c2b, float* logits)
{
  int i = blockIdx.x * 256 + threadIdx.x;   // 1280
  if (i >= 1280) return;
  int b = i / 10, c = i % 10;
  float s = c2b[c];
  for (int j = 0; j < 128; j++) s += hc[b * 128 + j] * c2w[j * 10 + c];
  logits[i] = s;
}

__global__ void k_logsoftmax(const float* logits, float* outp) {
  int b = blockIdx.x * 64 + threadIdx.x;
  if (b >= 128) return;
  float m = -1e30f;
  for (int c = 0; c < 10; c++) m = fmaxf(m, logits[b * 10 + c]);
  float s = 0.f;
  for (int c = 0; c < 10; c++) s += expf(logits[b * 10 + c] - m);
  float l = logf(s) + m;
  for (int c = 0; c < 10; c++) outp[b * 10 + c] = logits[b * 10 + c] - l;
}

// ---------------------------------------------------------------------------
// Host launcher.  Workspace layout (assumes ws_size >= ~264 MB).
// ---------------------------------------------------------------------------
extern "C" void kernel_launch(void* const* d_in, const int* in_sizes, int n_in,
                              void* d_out, int out_size, void* d_ws, size_t ws_size,
                              hipStream_t stream)
{
  const float* x       = (const float*)d_in[0];
  const int*   ei      = (const int*)d_in[1];
  const int*   src     = ei;
  const int*   dst     = ei + NEDGES;
  const float* enc_w   = (const float*)d_in[2];
  const float* enc_b   = (const float*)d_in[3];
  const float* gconv_w = (const float*)d_in[4];
  const float* gconv_b = (const float*)d_in[5];
  const float* wg1_w = (const float*)d_in[6];  const float* wg1_b = (const float*)d_in[7];
  const float* wg2_w = (const float*)d_in[8];  const float* wg2_b = (const float*)d_in[9];
  const float* pg1_w = (const float*)d_in[10]; const float* pg1_b = (const float*)d_in[11];
  const float* pg2_w = (const float*)d_in[12]; const float* pg2_b = (const float*)d_in[13];
  const float* q_w = (const float*)d_in[14];  const float* q_b = (const float*)d_in[15];
  const float* k_w = (const float*)d_in[16];  const float* k_b = (const float*)d_in[17];
  const float* v_w = (const float*)d_in[18];  const float* v_b = (const float*)d_in[19];
  const float* o_w = (const float*)d_in[20];  const float* o_b = (const float*)d_in[21];
  const float* ro_w = (const float*)d_in[22]; const float* ro_b = (const float*)d_in[23];
  const float* c1_w = (const float*)d_in[24]; const float* c1_b = (const float*)d_in[25];
  const float* c2_w = (const float*)d_in[26]; const float* c2_b = (const float*)d_in[27];
  (void)in_sizes; (void)n_in; (void)out_size; (void)ws_size;

  // ---- workspace carve-up ----
  float* fw    = (float*)d_ws;
  float* bufH  = fw;                 // h / K-gemm scatter scratch
  float* bufX1 = bufH  + NND;        // x1 / K fp32
  float* bufX2 = bufX1 + NND;        // x2 / V fp32
  float* bufX3 = bufX2 + NND;        // x3 / Q,O,O2
  float* bufT1 = bufX3 + NND;        // gcn gemm out / hc
  float* bufT2 = bufT1 + NND;        // gcn scatter accum / xf

  unsigned short* ub    = (unsigned short*)(bufT2 + NND);
  unsigned short* ub_x  = ub;                         // NND
  unsigned short* ub_K  = ub_x + NND;                 // NND
  unsigned short* ub_V  = ub_K + NND;                 // NND
  unsigned short* ub_xp = ub_V + NND;                 // PND
  unsigned short* ub_Ob = ub_xp + PND;                // PND
  unsigned short* ub_xg = ub_Ob + PND;                // 16384
  unsigned short* ub_hc = ub_xg + 16384;              // 16384 (unused slot)
  unsigned short* ub_W  = ub_hc + 16384;              // 15 * 16384

  float* sf    = (float*)(ub_W + 15 * 16384);
  float* deg   = sf;                float* dinv  = deg   + NNODES;
  float* s1    = dinv  + NNODES;    float* hv    = s1    + NNODES;
  float* s2    = hv    + NNODES;    float* scA   = s2    + NNODES;
  float* scB   = scA   + NNODES;    float* scC   = scB   + NNODES;
  float* score = scC   + NNODES;
  float* vals  = score + NNODES;                     // 16384
  float* logit = vals  + 16384;                      // 1280
  int*   idxb  = (int*)(logit + 1280);               // 16384

  float* Q  = bufX3;
  float* O  = bufX3 + PND;
  float* O2 = bufX3 + 2 * PND;
  float* outp  = (float*)d_out;       // [0,1280): logp
  float* outxg = outp + 1280;         // [1280, 1280+16384): xg

  const int GN   = (int)(NND / 256);      // 32768 blocks for n*128 elementwise
  const int GE   = NEDGES / 256;          // 2048
  const int GNn  = NNODES / 256;          // 256
  const int GMM  = NNODES / 128;          // 512 gemm blocks
  const int GMMP = (BGRAPH * RATIO) / 128; // 128

  // ---- weight prep: transpose + bf16 (slots: 0 enc, 1..9 gconv, 10 q, 11 k, 12 v, 13 o, 14 c1)
  k_transpose_bf16<<<64, 256, 0, stream>>>(enc_w, ub_W, 128, 128);
  for (int lk = 0; lk < 9; lk++)
    k_transpose_bf16<<<64, 256, 0, stream>>>(gconv_w + (size_t)lk * 16384, ub_W + (size_t)(1 + lk) * 16384, 128, 128);
  k_transpose_bf16<<<64, 256, 0, stream>>>(q_w,  ub_W + (size_t)10 * 16384, 128, 128);
  k_transpose_bf16<<<64, 256, 0, stream>>>(k_w,  ub_W + (size_t)11 * 16384, 128, 128);
  k_transpose_bf16<<<64, 256, 0, stream>>>(v_w,  ub_W + (size_t)12 * 16384, 128, 128);
  k_transpose_bf16<<<64, 256, 0, stream>>>(o_w,  ub_W + (size_t)13 * 16384, 128, 128);
  k_transpose_bf16<<<64, 256, 0, stream>>>(c1_w, ub_W + (size_t)14 * 16384, 128, 128);

  // ---- degree / normalization ----
  k_zero<<<GNn, 256, 0, stream>>>(deg, NNODES);
  k_deg<<<GE, 256, 0, stream>>>(dst, deg, NEDGES);
  k_dinv<<<GNn, 256, 0, stream>>>(deg, dinv, NNODES);

  // ---- encoder: h = x @ enc_w + enc_b ----
  k_f32_to_bf16<<<GN, 256, 0, stream>>>(x, ub_x, (int)NND);
  k_gemm128<<<GMM, 256, 0, stream>>>(ub_x, ub_W, enc_b, nullptr, bufH, 0);

  // ---- 3 mkgc layers, 3 GCN kernels each ----
  float* layer_in[3]  = {bufH, bufX1, bufX2};
  float* layer_acc[3] = {bufX1, bufX2, bufX3};
  for (int l = 0; l < 3; l++) {
    k_f32_to_bf16<<<GN, 256, 0, stream>>>(layer_in[l], ub_x, (int)NND);
    k_zero<<<GN, 256, 0, stream>>>(layer_acc[l], (int)NND);
    for (int k = 0; k < 3; k++) {
      int lk = l * 3 + k;
      k_gemm128<<<GMM, 256, 0, stream>>>(ub_x, ub_W + (size_t)(1 + lk) * 16384, nullptr, nullptr, bufT1, 0);
      k_zero<<<GN, 256, 0, stream>>>(bufT2, (int)NND);
      k_scatter128<<<NEDGES, 128, 0, stream>>>(bufT1, src, dst, dinv, bufT2);
      k_gcn_finalize<<<GN, 256, 0, stream>>>(bufT2, bufT1, dinv, gconv_b + lk * 128, layer_acc[l], 1);
    }
  }

  // ---- layer-attention scores + mix -> xf (bufT2) ----
  float* xs[3] = {bufX1, bufX2, bufX3};
  float* ss[3] = {scA, scB, scC};
  for (int l = 0; l < 3; l++) {
    k_dot2<<<NNODES / 8, 256, 0, stream>>>(xs[l], wg1_w, wg2_w, s1, hv, NNODES);
    k_zero<<<GNn, 256, 0, stream>>>(s2, NNODES);
    k_scatter1<<<GE, 256, 0, stream>>>(hv, src, dst, dinv, s2, NEDGES);
    k_gla_combine<<<GNn, 256, 0, stream>>>(s1, hv, s2, dinv, wg1_b, wg2_b, ss[l], NNODES);
  }
  k_mix3<<<GN, 256, 0, stream>>>(scA, scB, scC, bufX1, bufX2, bufX3, bufT2);
  float* xf = bufT2;

  // ---- pooling score, top-k, gather ----
  k_dot2<<<NNODES / 8, 256, 0, stream>>>(xf, pg1_w, pg2_w, s1, hv, NNODES);
  k_zero<<<GNn, 256, 0, stream>>>(s2, NNODES);
  k_scatter1<<<GE, 256, 0, stream>>>(hv, src, dst, dinv, s2, NEDGES);
  k_gla_combine<<<GNn, 256, 0, stream>>>(s1, hv, s2, dinv, pg1_b, pg2_b, score, NNODES);
  k_topk<<<BGRAPH, 512, 0, stream>>>(score, idxb, vals);
  k_gather_pool<<<BGRAPH * RATIO, 128, 0, stream>>>(xf, idxb, vals, ub_xp);

  // ---- K = gcn(xf, k_w, k_b), V = gcn(xf, v_w, v_b) ----
  k_f32_to_bf16<<<GN, 256, 0, stream>>>(xf, ub_x, (int)NND);
  k_gemm128<<<GMM, 256, 0, stream>>>(ub_x, ub_W + (size_t)11 * 16384, nullptr, nullptr, bufT1, 0);
  k_zero<<<GN, 256, 0, stream>>>(bufH, (int)NND);
  k_scatter128<<<NEDGES, 128, 0, stream>>>(bufT1, src, dst, dinv, bufH);
  k_gcn_finalize<<<GN, 256, 0, stream>>>(bufH, bufT1, dinv, k_b, bufX1, 0);
  k_f32_to_bf16<<<GN, 256, 0, stream>>>(bufX1, ub_K, (int)NND);

  k_gemm128<<<GMM, 256, 0, stream>>>(ub_x, ub_W + (size_t)12 * 16384, nullptr, nullptr, bufT1, 0);
  k_zero<<<GN, 256, 0, stream>>>(bufH, (int)NND);
  k_scatter128<<<NEDGES, 128, 0, stream>>>(bufT1, src, dst, dinv, bufH);
  k_gcn_finalize<<<GN, 256, 0, stream>>>(bufH, bufT1, dinv, v_b, bufX2, 0);
  k_f32_to_bf16<<<GN, 256, 0, stream>>>(bufX2, ub_V, (int)NND);

  // ---- Q = xp @ q_w + q_b ----
  k_gemm128<<<GMMP, 256, 0, stream>>>(ub_xp, ub_W + (size_t)10 * 16384, q_b, nullptr, Q, 0);

  // ---- attention: O = Q + softmax(QK^T/sqrt(D)) V ----
  k_attention<<<BGRAPH * HEADS * 8, 32, 0, stream>>>(Q, ub_K, ub_V, O);

  // ---- O2 = O + relu(O @ o_w + o_b) ----
  k_f32_to_bf16<<<(int)(PND / 256), 256, 0, stream>>>(O, ub_Ob, (int)PND);
  k_gemm128<<<GMMP, 256, 0, stream>>>(ub_Ob, ub_W + (size_t)13 * 16384, o_b, O, O2, 2);

  // ---- readout xg, classifier ----
  k_readout<<<64, 256, 0, stream>>>(O2, ro_w, ro_b, outxg, ub_xg);
  k_gemm128<<<1, 256, 0, stream>>>(ub_xg, ub_W + (size_t)14 * 16384, c1_b, nullptr, bufT1, 1);
  k_logits<<<5, 256, 0, stream>>>(bufT1, c2_w, c2_b, logit);
  k_logsoftmax<<<2, 64, 0, stream>>>(logit, outp);
}
